// MultiModalLSTM_69552700392067
// MI455X (gfx1250) — matrix operand order repaired and checked by reading, not compiled, not run on hardware
//
#include <hip/hip_runtime.h>
#include <hip/hip_bf16.h>

// ---------------------------------------------------------------------------
// MultiModal LSTM for MI455X (gfx1250, wave32, WMMA bf16)
//
//   gates = [h*d | x_t] @ [W_hh | W_ih]^T  + (b_ih+b_hh)     K = 1024+512
//
// Strategy:
//   * One-time pack: [W_hh|W_ih] -> bf16 WMMA B-fragments (12 MB ws)
//                    x           -> bf16 WMMA A-fragments (32 MB ws)
//   * d = img @ W_m^T (small VALU GEMM, 0.27 GFLOP)
//   * Persistent recurrent kernel: 64 WGs x 256 threads (8 waves).
//     - WG ht owns hidden units [16*ht,16*ht+16): gate rows {ht,64+ht,128+ht,192+ht}*16
//     - 192 KB weight slice loaded ONCE into LDS via the Tensor Data Mover
//       (tensor_load_to_lds + s_wait_tensorcnt), reused for all 512 steps.
//     - h exchanged between WGs as pre-gated bf16 A-fragments in L2
//       (double buffered), device-wide barrier (atomic ctr + s_sleep) per step.
//     - c and d slices resident in LDS, biases in LDS.
// ---------------------------------------------------------------------------

typedef __bf16  v16bf __attribute__((ext_vector_type(16)));
typedef float   v8f   __attribute__((ext_vector_type(8)));
typedef unsigned int  tdm_u32x4 __attribute__((ext_vector_type(4)));
typedef int           tdm_i32x8 __attribute__((ext_vector_type(8)));
typedef int           tdm_i32x4 __attribute__((ext_vector_type(4)));

#define TSTEPS 512
#define BATCH  64
#define HID    1024
#define DIN    512
#define GROWS  4096          // 4*HID
#define NKT    48            // Kcat=1536 / 32
#define NKT_H  32            // hidden part
#define NKT_X  16            // input part
#define NWG    64            // one WG per 16-hidden-unit slice
#define FRAG   512           // bf16 elems per 16x32 / 32x16 fragment (1 KB)

// workspace layout (bytes)
#define WCAT_OFF   0ull                                     // 256 ntiles*48*512*2 = 12 MB
#define WCAT_BYTES (256ull*NKT*FRAG*2)
#define XFRAG_OFF  (WCAT_OFF + WCAT_BYTES)                  // 512 t *4*16*512*2 = 32 MB
#define XFRAG_BYTES (512ull*4*NKT_X*FRAG*2)
#define HGFRAG_OFF (XFRAG_OFF + XFRAG_BYTES)                // 2 bufs *4*32*512*2 = 256 KB
#define HGFRAG_BYTES (2ull*4*NKT_H*FRAG*2)
#define DGATE_OFF  (HGFRAG_OFF + HGFRAG_BYTES)              // 64*1024 f32
#define DGATE_BYTES (64ull*HID*4)
#define BSUM_OFF   (DGATE_OFF + DGATE_BYTES)                // 4096 f32
#define BSUM_BYTES (GROWS*4ull)
#define CTR_OFF    (BSUM_OFF + BSUM_BYTES)                  // sync counter
#define CTR_BYTES  256ull

// dynamic LDS partition for the persistent kernel
#define WLDS_BYTES (4*NKT*FRAG*2)        // 196608: weight fragments
#define GLDS_BYTES (4*BATCH*16*4)        // 16384 : gate pre-activations
#define CLDS_BYTES (BATCH*16*4)          // 4096  : cell state slice
#define DLDS_BYTES (BATCH*16*4)          // 4096  : multimodal gate slice
#define BLDS_BYTES (4*16*4)              // 256   : bias slice
#define SMEM_BYTES (WLDS_BYTES+GLDS_BYTES+CLDS_BYTES+DLDS_BYTES+BLDS_BYTES)

// A-fragment index helpers (ISA 7.12.2, 16-bit A 16x32):
//   half = (K>>3)&1 ; lane = half*16 + M ; j = (K&7) | ((K>>4)<<3)
__device__ __forceinline__ int afrag_lane(int M, int K) { return (((K >> 3) & 1) << 4) | M; }
__device__ __forceinline__ int afrag_j(int K)           { return (K & 7) | ((K >> 4) << 3); }

// ---------------------------------------------------------------------------
// Pack [W_hh | W_ih] into bf16 B-fragments: wcat[ntile][kt][lane*16+j]
// B layout (16-bit 32x16): lane: N = lane%16, half = lane/16; K = 8*half + (j&7) + 16*(j>>3)
// ---------------------------------------------------------------------------
__global__ void pack_wcat_kernel(const float* __restrict__ Whh,
                                 const float* __restrict__ Wih,
                                 __bf16* __restrict__ wcat) {
    int fragid = blockIdx.x * 8 + (threadIdx.x >> 5);   // 0..12287
    int lane   = threadIdx.x & 31;
    int ntile  = fragid / NKT;
    int kt     = fragid % NKT;
    int N = lane & 15, half = lane >> 4;
    int n = ntile * 16 + N;                             // gate row
    __bf16* dst = wcat + (size_t)fragid * FRAG + lane * 16;
#pragma unroll
    for (int j = 0; j < 16; ++j) {
        int K = 8 * half + (j & 7) + ((j >> 3) << 4);
        int k = kt * 32 + K;                            // concatenated K
        float v = (k < HID) ? Whh[(size_t)n * HID + k]
                            : Wih[(size_t)n * DIN + (k - HID)];
        dst[j] = (__bf16)v;
    }
}

// ---------------------------------------------------------------------------
// Pack x[b][t][dk] into bf16 A-fragments: xfrag[t][mtile][kt'][...]
// ---------------------------------------------------------------------------
__global__ void pack_x_kernel(const float* __restrict__ x,
                              __bf16* __restrict__ xfrag) {
    size_t idx = (size_t)blockIdx.x * 256 + threadIdx.x;   // exactly B*T*D threads
    int dk = (int)(idx % DIN);
    int t  = (int)((idx / DIN) % TSTEPS);
    int b  = (int)(idx / ((size_t)DIN * TSTEPS));
    float v = x[idx];
    int ktp = dk >> 5, K = dk & 31;
    int ln  = afrag_lane(b & 15, K);
    int mt  = b >> 4;
    xfrag[(((size_t)t * 4 + mt) * NKT_X + ktp) * FRAG + ln * 16 + afrag_j(K)] = (__bf16)v;
}

__global__ void bias_sum_kernel(const float* __restrict__ bih,
                                const float* __restrict__ bhh,
                                float* __restrict__ bsum) {
    int i = blockIdx.x * 256 + threadIdx.x;
    if (i < GROWS) bsum[i] = bih[i] + bhh[i];
}

// d[b][h] = dot(img[b,:], W_m[h,:]) over IMG=2048  (0.27 GFLOP, VALU is fine)
__global__ void dgate_kernel(const float* __restrict__ img,
                             const float* __restrict__ Wm,
                             float* __restrict__ d) {
    int h = blockIdx.x * 256 + threadIdx.x;   // < 1024
    int b = blockIdx.y;
    const float* ir = img + (size_t)b * 2048;
    const float* wr = Wm  + (size_t)h * 2048;
    float acc = 0.f;
#pragma unroll 8
    for (int k = 0; k < 2048; ++k) acc = fmaf(ir[k], wr[k], acc);
    d[b * HID + h] = acc;
}

// ---------------------------------------------------------------------------
// Persistent recurrent kernel. grid = 64 WGs (must be co-resident: 1 WG/WGP
// at ~221 KB LDS each), block = 256 (8 wave32).
// ---------------------------------------------------------------------------
__global__ __launch_bounds__(256, 1)
void lstm_persistent_kernel(const float* __restrict__ dgate,
                            const float* __restrict__ bsum,
                            const __bf16* __restrict__ wcat,
                            const __bf16* __restrict__ xfrag,
                            __bf16* __restrict__ hgfrag,
                            unsigned* __restrict__ ctr,
                            float* __restrict__ out,
                            float* __restrict__ hn,
                            float* __restrict__ cn) {
    extern __shared__ char smem[];
    __bf16* wlds = (__bf16*)smem;                               // [4][48][512]
    float*  glds = (float*)(smem + WLDS_BYTES);                 // [4][64][16]
    float*  clds = glds + 4 * BATCH * 16;                       // [64][16]
    float*  dlds = clds + BATCH * 16;                           // [64][16]
    float*  blds = dlds + BATCH * 16;                           // [4][16]

    const int ht   = blockIdx.x;        // hidden slice 0..63
    const int h0   = ht * 16;
    const int tid  = threadIdx.x;
    const int lane = tid & 31;
    const int wave = tid >> 5;

    // ---- one-time: TDM weight slice -> LDS (4 x 48 KB contiguous copies) ----
    if (tid == 0) {
        for (int g = 0; g < 4; ++g) {
            int ntile = g * 64 + ht;
            unsigned long long gaddr =
                (unsigned long long)(const void*)(wcat + (size_t)ntile * NKT * FRAG);
            unsigned ldsoff = (unsigned)(unsigned long long)(void*)(wlds + g * NKT * FRAG);
            const unsigned elems = NKT * FRAG * 2 / 8;          // 6144 x 8B = 48 KB
            tdm_u32x4 d0;                                       // D# group 0
            d0.x = 1u;                                          // count=1
            d0.y = ldsoff;                                      // lds_addr
            d0.z = (unsigned)(gaddr & 0xffffffffu);             // global_addr[31:0]
            d0.w = (unsigned)((gaddr >> 32) & 0x01ffffffu) | (2u << 30); // addr[56:32] | type=2
            tdm_i32x8 d1;                                       // D# group 1
            d1[0] = 3 << 16;                                    // data_size = 8B, mask=0
            d1[1] = (int)(elems << 16);                         // tensor_dim0 lo16 @ bits 63:48
            d1[2] = 0;                                          // tensor_dim0 hi16 / dim1 lo
            d1[3] = (int)(elems << 16);                         // tile_dim0 @ bits 127:112
            d1[4] = 0;                                          // tile_dim1/2 unused
            d1[5] = (int)elems;                                 // tensor_dim0_stride lo32
            d1[6] = 0; d1[7] = 0;
            tdm_i32x4 dz4 = {0, 0, 0, 0};                       // D# group 2 (unused)
            tdm_i32x4 dz4b = {0, 0, 0, 0};                      // D# group 3 (unused)
            tdm_i32x8 dz8 = {0, 0, 0, 0, 0, 0, 0, 0};           // extra group (6-arg form)
            __builtin_amdgcn_tensor_load_to_lds(d0, d1, dz4, dz4b, dz8, 0);
        }
        __builtin_amdgcn_s_wait_tensorcnt(0);
    }
    // d slice, bias slice, c init
    for (int i = tid; i < BATCH * 16; i += 256) {
        int b = i >> 4, j = i & 15;
        dlds[i] = dgate[b * HID + h0 + j];
        clds[i] = 0.f;
    }
    for (int i = tid; i < 64; i += 256) {
        int g = i >> 4, j = i & 15;
        blds[i] = bsum[g * HID + h0 + j];
    }
    __syncthreads();

    // wave -> 2 C tiles (same mtile, adjacent gate groups)
    const int mtile = wave >> 1;
    const int ga    = (2 * wave) & 3;
    const int gb    = ga + 1;

    for (int t = 0; t < TSTEPS; ++t) {
        v8f acc0 = {}; v8f acc1 = {};
        const __bf16* hgsrc = hgfrag + (size_t)(t & 1) * (4 * NKT_H * FRAG);
        const __bf16* xsrc  = xfrag  + (size_t)t * (4 * NKT_X * FRAG);

        // hidden K range: A from gated-h fragments (produced last step)
#pragma unroll 4
        for (int kt = 0; kt < NKT_H; ++kt) {
            v16bf a  = *(const v16bf*)(hgsrc + ((size_t)(mtile * NKT_H + kt)) * FRAG + lane * 16);
            v16bf b0 = *(const v16bf*)(wlds + (ga * NKT + kt) * FRAG + lane * 16);
            v16bf b1 = *(const v16bf*)(wlds + (gb * NKT + kt) * FRAG + lane * 16);
            acc0 = __builtin_amdgcn_wmma_f32_16x16x32_bf16(false, a, false, b0, (short)0, acc0, false, false);
            acc1 = __builtin_amdgcn_wmma_f32_16x16x32_bf16(false, a, false, b1, (short)0, acc1, false, false);
        }
        // input K range: A from pre-packed x fragments
#pragma unroll 4
        for (int kt = 0; kt < NKT_X; ++kt) {
            v16bf a  = *(const v16bf*)(xsrc + ((size_t)(mtile * NKT_X + kt)) * FRAG + lane * 16);
            v16bf b0 = *(const v16bf*)(wlds + (ga * NKT + NKT_H + kt) * FRAG + lane * 16);
            v16bf b1 = *(const v16bf*)(wlds + (gb * NKT + NKT_H + kt) * FRAG + lane * 16);
            acc0 = __builtin_amdgcn_wmma_f32_16x16x32_bf16(false, a, false, b0, (short)0, acc0, false, false);
            acc1 = __builtin_amdgcn_wmma_f32_16x16x32_bf16(false, a, false, b1, (short)0, acc1, false, false);
        }

        // spill accumulators: C/D layout lane N = lane%16, row M = v + 8*(lane/16)
        {
            int N = lane & 15, halfm = lane >> 4;
#pragma unroll
            for (int v = 0; v < 8; ++v) {
                int b = mtile * 16 + v + 8 * halfm;
                glds[(ga * BATCH + b) * 16 + N] = acc0[v];
                glds[(gb * BATCH + b) * 16 + N] = acc1[v];
            }
        }
        __syncthreads();

        // pointwise LSTM cell for this hidden slice (1024 units / 256 threads)
        for (int u = tid; u < BATCH * 16; u += 256) {
            int b = u >> 4, j = u & 15;
            float gi = glds[(0 * BATCH + b) * 16 + j] + blds[0 * 16 + j];
            float gf = glds[(1 * BATCH + b) * 16 + j] + blds[1 * 16 + j];
            float gg = glds[(2 * BATCH + b) * 16 + j] + blds[2 * 16 + j];
            float go = glds[(3 * BATCH + b) * 16 + j] + blds[3 * 16 + j];
            float i_ = 1.f / (1.f + __expf(-gi));
            float f_ = 1.f / (1.f + __expf(-gf));
            float g_ = tanhf(gg);
            float o_ = 1.f / (1.f + __expf(-go));
            float c  = f_ * clds[u] + i_ * g_;
            clds[u]  = c;
            float h  = o_ * tanhf(c);
            out[((size_t)b * TSTEPS + t) * HID + h0 + j] = h;
            if (t == TSTEPS - 1) hn[b * HID + h0 + j] = h;
            // gated h -> next step's A fragment (other buffer)
            float hg = h * dlds[u];
            int k = h0 + j, ktile = k >> 5, K = k & 31;
            int ln = afrag_lane(b & 15, K), mt = b >> 4;
            hgfrag[(size_t)((t + 1) & 1) * (4 * NKT_H * FRAG)
                   + ((size_t)(mt * NKT_H + ktile)) * FRAG + ln * 16 + afrag_j(K)] = (__bf16)hg;
        }
        __threadfence();          // publish hg fragments device-wide
        __syncthreads();
        if (tid == 0) {           // device-wide barrier: arrive + sleep-spin
            __hip_atomic_fetch_add(ctr, 1u, __ATOMIC_ACQ_REL, __HIP_MEMORY_SCOPE_AGENT);
            unsigned tgt = (unsigned)(t + 1) * gridDim.x;
            while (__hip_atomic_load(ctr, __ATOMIC_ACQUIRE, __HIP_MEMORY_SCOPE_AGENT) < tgt)
                __builtin_amdgcn_s_sleep(4);
        }
        __syncthreads();
    }

    for (int u = tid; u < BATCH * 16; u += 256) {
        int b = u >> 4, j = u & 15;
        cn[b * HID + h0 + j] = clds[u];
    }
}

// ---------------------------------------------------------------------------
extern "C" void kernel_launch(void* const* d_in, const int* in_sizes, int n_in,
                              void* d_out, int out_size, void* d_ws, size_t ws_size,
                              hipStream_t stream) {
    const float* x   = (const float*)d_in[0];   // [B,T,D]
    const float* img = (const float*)d_in[1];   // [B,IMG]
    const float* Wih = (const float*)d_in[2];   // [4H,D]
    const float* Whh = (const float*)d_in[3];   // [4H,H]
    const float* bih = (const float*)d_in[4];   // [4H]
    const float* bhh = (const float*)d_in[5];   // [4H]
    const float* Wm  = (const float*)d_in[6];   // [H,IMG]

    char* ws = (char*)d_ws;
    __bf16*   wcat   = (__bf16*)(ws + WCAT_OFF);
    __bf16*   xfrag  = (__bf16*)(ws + XFRAG_OFF);
    __bf16*   hgfrag = (__bf16*)(ws + HGFRAG_OFF);
    float*    dgatep = (float*)(ws + DGATE_OFF);
    float*    bsum   = (float*)(ws + BSUM_OFF);
    unsigned* ctr    = (unsigned*)(ws + CTR_OFF);

    float* out = (float*)d_out;                      // [B,T,H]
    float* hn  = out + (size_t)BATCH * TSTEPS * HID; // [B,H]
    float* cn  = hn + (size_t)BATCH * HID;           // [B,H]

    // h_0 fragments must be zero; barrier counter must start at 0 every call
    (void)hipMemsetAsync(ws + HGFRAG_OFF, 0, HGFRAG_BYTES, stream);
    (void)hipMemsetAsync(ws + CTR_OFF, 0, CTR_BYTES, stream);

    (void)hipFuncSetAttribute((const void*)lstm_persistent_kernel,
                              hipFuncAttributeMaxDynamicSharedMemorySize,
                              (int)SMEM_BYTES);

    pack_wcat_kernel<<<256 * NKT / 8, 256, 0, stream>>>(Whh, Wih, wcat);
    pack_x_kernel<<<(BATCH * TSTEPS * DIN) / 256, 256, 0, stream>>>(x, xfrag);
    bias_sum_kernel<<<GROWS / 256, 256, 0, stream>>>(bih, bhh, bsum);
    dgate_kernel<<<dim3(HID / 256, BATCH), 256, 0, stream>>>(img, Wm, dgatep);

    lstm_persistent_kernel<<<NWG, 256, SMEM_BYTES, stream>>>(
        dgatep, bsum, wcat, xfrag, hgfrag, ctr, out, hn, cn);
}
